// GNN_RNN_HiddenStatePredictor_7198365188341
// MI455X (gfx1250) — matrix operand order repaired and checked by reading, not compiled, not run on hardware
//
#include <hip/hip_runtime.h>
#include <math.h>

// Problem dims
#define TDIM 32
#define BDIM 16
#define VDIM 128
#define DIN_ 64
#define HDIM 128
#define HN_  32
#define DOUT_ 64

typedef __attribute__((ext_vector_type(16))) _Float16 v16h;
typedef __attribute__((ext_vector_type(8)))  _Float16 v8h;
typedef __attribute__((ext_vector_type(8)))  float    v8f;
typedef __attribute__((ext_vector_type(4)))  unsigned int u32x4;

union V16HU { u32x4 u[2]; v16h h; };
union V16HP { v8h p[2]; v16h h; };

__device__ __forceinline__ v8f wmma16(v16h a, v16h b, v8f c) {
  return __builtin_amdgcn_wmma_f32_16x16x32_f16(false, a, false, b, (short)0, c, false, false);
}

__device__ __forceinline__ v8f zero8() {
  v8f c;
#pragma unroll
  for (int r = 0; r < 8; ++r) c[r] = 0.0f;
  return c;
}

__device__ __forceinline__ v8h zero8h() {
  v8h c;
#pragma unroll
  for (int r = 0; r < 8; ++r) c[r] = (_Float16)0.0f;
  return c;
}

// B-fragment (16x16x32 f16): lane = N (mod 16); lanes 0-15 hold K {0..7,16..23},
// lanes 16-31 hold K {8..15,24..31}; source is [n][k] f16, k contiguous.
__device__ __forceinline__ v16h bfrag_nk(const _Float16* np, int kt, int hi) {
  V16HU u;
  const u32x4* p = (const u32x4*)(np + kt * 32 + (hi ? 8 : 0));
  u.u[0] = p[0]; u.u[1] = p[2];
  return u.h;
}

// A-fragment (16x16x32 f16) from a row-major f16 row pointer:
// lanes 0-15: K {0..7,16..23}; lanes 16-31: K {8..15,24..31}.
__device__ __forceinline__ v16h afrag_row(const _Float16* rowp, int kt, int hi) {
  V16HU u;
  const u32x4* p = (const u32x4*)(rowp + kt * 32 + (hi ? 8 : 0));
  u.u[0] = p[0];
  u.u[1] = p[2];   // +16 halves
  return u.h;
}

// ---------------------------------------------------------------- weight pack
__global__ __launch_bounds__(256) void k_pack(
    const float* __restrict__ Wp, const float* __restrict__ W1,
    const float* __restrict__ W2, const float* __restrict__ Wn1,
    const float* __restrict__ Wn2, const float* __restrict__ Wih,
    const float* __restrict__ Whh, const float* __restrict__ Wo,
    _Float16* __restrict__ wpT, _Float16* __restrict__ w1aT,
    _Float16* __restrict__ w1bT, _Float16* __restrict__ w2T,
    _Float16* __restrict__ wn1T, _Float16* __restrict__ wn2T,
    _Float16* __restrict__ wih16, _Float16* __restrict__ whh16,
    _Float16* __restrict__ woT)
{
  int i = blockIdx.x * 256 + threadIdx.x;
  if (i < 8192) {            // Wp: (64,128) -> wpT[n*64+k]
    int n = i >> 6, k = i & 63;
    wpT[i] = (_Float16)Wp[k * HDIM + n];
  }
  if (i < 8192) {            // Wo: (128,64) -> woT[n*128+k]
    int n = i >> 7, k = i & 127;
    woT[i] = (_Float16)Wo[k * DOUT_ + n];
  }
  if (i < 16384) {           // 128x128 transposes
    int n = i >> 7, k = i & 127;
    w1aT[i] = (_Float16)W1[k * HDIM + n];
    w1bT[i] = (_Float16)W1[(HDIM + k) * HDIM + n];
    w2T[i]  = (_Float16)W2[k * HDIM + n];
    wn1T[i] = (_Float16)Wn1[k * HDIM + n];
    wn2T[i] = (_Float16)Wn2[k * HDIM + n];
  }
  if (i < 49152) {           // GRU weights: row-major Wih[n][k] already is B layout
    wih16[i] = (_Float16)Wih[i];
    whh16[i] = (_Float16)Whh[i];
  }
}

// ------------------------------------------------------- h = x_t @ Wp + bp (f16)
__global__ __launch_bounds__(32) void k_proj(
    const float* __restrict__ x, const float* __restrict__ bp,
    const _Float16* __restrict__ wpT, _Float16* __restrict__ h16)
{
  const int blk = blockIdx.x;         // tb*8 + vtile
  const int vt = blk & 7;
  const int tb = blk >> 3;            // t*16 + b
  const int t = tb >> 4;
  const int b = tb & 15;
  const int lane = threadIdx.x & 31;
  const int hi = (lane >> 4) & 1;
  const int m = lane & 15;
  const int v0 = vt * 16;

  v16h af[2];
#pragma unroll
  for (int kt = 0; kt < 2; ++kt) {
    v16h a;
#pragma unroll
    for (int e = 0; e < 16; ++e) {
      int k = kt * 32 + ((e < 8) ? e : e + 8) + (hi ? 8 : 0);
      a[e] = (_Float16)x[(((size_t)(b * VDIM + v0 + m)) * TDIM + t) * DIN_ + k];
    }
    af[kt] = a;
  }
#pragma unroll
  for (int nt = 0; nt < 8; ++nt) {
    v8f c = zero8();
    const int nabs = nt * 16 + m;
#pragma unroll
    for (int kt = 0; kt < 2; ++kt)
      c = wmma16(af[kt], bfrag_nk(wpT + (size_t)nabs * DIN_, kt, hi), c);
    float bv = bp[nabs];
#pragma unroll
    for (int r = 0; r < 8; ++r) {
      int mr = r + (hi ? 8 : 0);
      h16[((size_t)tb * VDIM + v0 + mr) * HDIM + nabs] = (_Float16)(c[r] + bv);
    }
  }
}

// -------------------------------------- ai = h@W1a + b1 (folded), bj = h@W1b
__global__ __launch_bounds__(64) void k_aibj(
    const _Float16* __restrict__ h16, const _Float16* __restrict__ w1aT,
    const _Float16* __restrict__ w1bT, const float* __restrict__ b1,
    _Float16* __restrict__ ai16, _Float16* __restrict__ bj16)
{
  const int blk = blockIdx.x;
  const int vt = blk & 7;
  const int tb = blk >> 3;
  const int wave = threadIdx.x >> 5;
  const int lane = threadIdx.x & 31;
  const int hi = (lane >> 4) & 1;
  const int m = lane & 15;
  const size_t rowbase = (size_t)tb * VDIM + vt * 16;

  const _Float16* W = wave ? w1bT : w1aT;
  _Float16* dst = wave ? bj16 : ai16;

  v16h af[4];
#pragma unroll
  for (int kt = 0; kt < 4; ++kt)
    af[kt] = afrag_row(h16 + (rowbase + m) * HDIM, kt, hi);

#pragma unroll
  for (int nt = 0; nt < 8; ++nt) {
    v8f c = zero8();
    const int nabs = nt * 16 + m;
#pragma unroll
    for (int kt = 0; kt < 4; ++kt)
      c = wmma16(af[kt], bfrag_nk(W + (size_t)nabs * HDIM, kt, hi), c);
    float bv = wave ? 0.0f : b1[nabs];
#pragma unroll
    for (int r = 0; r < 8; ++r) {
      int mr = r + (hi ? 8 : 0);
      dst[(rowbase + mr) * HDIM + nabs] = (_Float16)(c[r] + bv);
    }
  }
}

// --------- hot kernel: agg[i] = sum_j adj[i,j] * relu(relu(ai_i+bj_j)@W2+b2)
__global__ __launch_bounds__(128) void k_edge(
    const _Float16* __restrict__ ai16, const _Float16* __restrict__ bj16,
    const float* __restrict__ adj, const _Float16* __restrict__ w2T,
    const float* __restrict__ b2, _Float16* __restrict__ agg16)
{
  __shared__ alignas(16) _Float16 sW2t[HDIM * HDIM];  // 32 KB, [n][k]
  __shared__ alignas(16) _Float16 sA16[16 * HDIM];    // ai tile (b1 folded), 4 KB
  __shared__ alignas(16) _Float16 sBj16[16 * HDIM];   // bj tile, 4 KB
  __shared__ float sAdj[16 * 16];
  __shared__ float sB2[HDIM];

  const int blk = blockIdx.x;         // tb*8 + itile
  const int itile = blk & 7;
  const int tb = blk >> 3;
  const int b = tb & 15;
  const int i0 = itile * 16;
  const int tid = threadIdx.x;
  const int wave = tid >> 5;
  const int lane = tid & 31;
  const int hi = (lane >> 4) & 1;
  const int m = lane & 15;

  {
    const u32x4* src = (const u32x4*)w2T;
    u32x4* dst = (u32x4*)sW2t;
    for (int i = tid; i < (HDIM * HDIM) / 8; i += 128) dst[i] = src[i];
  }
  if (tid < HDIM) sB2[tid] = b2[tid];
  {
    // ai tile is 16 consecutive rows -> contiguous 4 KB copy
    const u32x4* src = (const u32x4*)(ai16 + ((size_t)tb * VDIM + i0) * HDIM);
    u32x4* dst = (u32x4*)sA16;
    for (int i = tid; i < 256; i += 128) dst[i] = src[i];
  }

  const v8h zh = zero8h();
  const v8f zf = zero8();

  v8f aggLoc[8];
#pragma unroll
  for (int nt = 0; nt < 8; ++nt) aggLoc[nt] = zf;

  for (int j0 = 0; j0 < VDIM; j0 += 16) {
    __syncthreads();
    {
      const u32x4* src = (const u32x4*)(bj16 + ((size_t)tb * VDIM + j0) * HDIM);
      u32x4* dst = (u32x4*)sBj16;
      for (int i = tid; i < 256; i += 128) dst[i] = src[i];
    }
    for (int i = tid; i < 256; i += 128) {
      int mm = i >> 4, jj = i & 15;
      sAdj[i] = adj[((size_t)b * VDIM + i0 + mm) * VDIM + (j0 + jj)];
    }
    if (j0 + 16 < VDIM) {   // prefetch next bj tile (global_prefetch_b8)
      const char* nxt = (const char*)(bj16 + ((size_t)tb * VDIM + j0 + 16) * HDIM);
      __builtin_prefetch(nxt + tid * 32, 0, 1);
    }
    __syncthreads();

    for (int jw = 0; jw < 2; ++jw) {
      const int jj = wave * 4 + jw * 2;   // this wave's j pair: jj, jj+1
      // Build A fragments of E = relu(ai_i + bj_j) with packed f16 math
      v16h afr0[4], afr1[4];
#pragma unroll
      for (int kt = 0; kt < 4; ++kt) {
        const v8h* pa = (const v8h*)(sA16 + m * HDIM + kt * 32 + (hi ? 8 : 0));
        const v8h* p0 = (const v8h*)(sBj16 + jj * HDIM + kt * 32 + (hi ? 8 : 0));
        const v8h* p1 = (const v8h*)(sBj16 + (jj + 1) * HDIM + kt * 32 + (hi ? 8 : 0));
        v8h a0 = pa[0], a1 = pa[2];       // K runs {base..+7} and {base+16..+23}
        V16HP u0, u1;
        u0.p[0] = __builtin_elementwise_max(a0 + p0[0], zh);
        u0.p[1] = __builtin_elementwise_max(a1 + p0[2], zh);
        u1.p[0] = __builtin_elementwise_max(a0 + p1[0], zh);
        u1.p[1] = __builtin_elementwise_max(a1 + p1[2], zh);
        afr0[kt] = u0.h;
        afr1[kt] = u1.h;
      }
      v8f adj0, adj1;
#pragma unroll
      for (int r = 0; r < 8; ++r) {
        adj0[r] = sAdj[(r + (hi ? 8 : 0)) * 16 + jj];
        adj1[r] = sAdj[(r + (hi ? 8 : 0)) * 16 + jj + 1];
      }

#pragma unroll
      for (int nt = 0; nt < 8; ++nt) {
        v8f c0 = zf, c1 = zf;
        const int nabs = nt * 16 + m;
#pragma unroll
        for (int kt = 0; kt < 4; ++kt) {
          v16h bf = bfrag_nk(sW2t + (size_t)nabs * HDIM, kt, hi);
          c0 = wmma16(afr0[kt], bf, c0);
          c1 = wmma16(afr1[kt], bf, c1);
        }
        const float bv = sB2[nabs];
        v8f v0 = __builtin_elementwise_max(c0 + bv, zf);  // V_PK_ADD_F32 + max
        v8f v1 = __builtin_elementwise_max(c1 + bv, zf);
        aggLoc[nt] = aggLoc[nt] + adj0 * v0 + adj1 * v1;  // V_PK_FMA_F32
      }
    }
  }

  // Deterministic cross-wave reduction (reuse sW2t as 4x2048 f32 scratch)
  __syncthreads();
  float* red = (float*)sW2t;
#pragma unroll
  for (int nt = 0; nt < 8; ++nt)
#pragma unroll
    for (int r = 0; r < 8; ++r) {
      int mr = r + (hi ? 8 : 0);
      red[wave * 2048 + mr * HDIM + nt * 16 + m] = aggLoc[nt][r];
    }
  __syncthreads();
  for (int i = tid; i < 2048; i += 128) {
    float s = red[i] + red[2048 + i] + red[4096 + i] + red[6144 + i];
    int mm = i >> 7, n = i & 127;
    agg16[((size_t)tb * VDIM + i0 + mm) * HDIM + n] = (_Float16)s;
  }
}

// ------------------------------------------- tmp = relu(agg @ Wn1 + bn1)
__global__ __launch_bounds__(32) void k_mlp1(
    const _Float16* __restrict__ src, const _Float16* __restrict__ wT,
    const float* __restrict__ bias, _Float16* __restrict__ dst)
{
  const int blk = blockIdx.x;
  const int vt = blk & 7;
  const int tb = blk >> 3;
  const int lane = threadIdx.x & 31;
  const int hi = (lane >> 4) & 1;
  const int m = lane & 15;
  const size_t rowbase = (size_t)tb * VDIM + vt * 16;

  v16h af[4];
#pragma unroll
  for (int kt = 0; kt < 4; ++kt)
    af[kt] = afrag_row(src + (rowbase + m) * HDIM, kt, hi);
#pragma unroll
  for (int nt = 0; nt < 8; ++nt) {
    v8f c = zero8();
    const int nabs = nt * 16 + m;
#pragma unroll
    for (int kt = 0; kt < 4; ++kt)
      c = wmma16(af[kt], bfrag_nk(wT + (size_t)nabs * HDIM, kt, hi), c);
    float bv = bias[nabs];
#pragma unroll
    for (int r = 0; r < 8; ++r) {
      int mr = r + (hi ? 8 : 0);
      float v = c[r] + bv;
      dst[(rowbase + mr) * HDIM + nabs] = (_Float16)(v > 0.0f ? v : 0.0f);
    }
  }
}

// ------------------ m[t,b] = mean_v relu(tmp @ Wn2 + bn2); one block per (t,b)
__global__ __launch_bounds__(32) void k_mlp2_mean(
    const _Float16* __restrict__ src, const _Float16* __restrict__ wT,
    const float* __restrict__ bias, float* __restrict__ mbuf)
{
  __shared__ float sAcc[HDIM];
  const int tb = blockIdx.x;          // 512
  const int lane = threadIdx.x & 31;
  const int hi = (lane >> 4) & 1;
  const int m = lane & 15;

  float part[8];
#pragma unroll
  for (int i = 0; i < 8; ++i) part[i] = 0.0f;

  for (int vt = 0; vt < 8; ++vt) {
    const size_t rowbase = (size_t)tb * VDIM + vt * 16;
    v16h af[4];
#pragma unroll
    for (int kt = 0; kt < 4; ++kt)
      af[kt] = afrag_row(src + (rowbase + m) * HDIM, kt, hi);
#pragma unroll
    for (int nt = 0; nt < 8; ++nt) {
      v8f c = zero8();
      const int nabs = nt * 16 + m;
#pragma unroll
      for (int kt = 0; kt < 4; ++kt)
        c = wmma16(af[kt], bfrag_nk(wT + (size_t)nabs * HDIM, kt, hi), c);
      float bv = bias[nabs];
      float s = 0.0f;
#pragma unroll
      for (int r = 0; r < 8; ++r) {
        float v = c[r] + bv;
        s += (v > 0.0f ? v : 0.0f);
      }
      part[nt] += s;
    }
  }
  for (int i = threadIdx.x; i < HDIM; i += 32) sAcc[i] = 0.0f;
  __syncthreads();
#pragma unroll
  for (int nt = 0; nt < 8; ++nt) atomicAdd(&sAcc[nt * 16 + m], part[nt]);
  __syncthreads();
  for (int i = threadIdx.x; i < HDIM; i += 32)
    mbuf[(size_t)tb * HDIM + i] = sAcc[i] * (1.0f / VDIM);
}

// ------------------------- cumsum over t + GRU (batch 16), persistent block
__global__ __launch_bounds__(256) void k_gru(
    const float* __restrict__ mbuf, const _Float16* __restrict__ wih16,
    const _Float16* __restrict__ whh16, const float* __restrict__ bih,
    const float* __restrict__ bhh, _Float16* __restrict__ rnn16)
{
  __shared__ float sCum[BDIM * HDIM];
  __shared__ float sH[BDIM * HDIM];
  __shared__ alignas(16) _Float16 sX16[BDIM * HDIM];
  __shared__ alignas(16) _Float16 sH16[BDIM * HDIM];
  __shared__ _Float16 sGi[BDIM * 3 * HDIM];
  __shared__ _Float16 sGh[BDIM * 3 * HDIM];

  const int tid = threadIdx.x;
  const int wave = tid >> 5;
  const int lane = tid & 31;
  const int hi = (lane >> 4) & 1;
  const int m = lane & 15;

  for (int i = tid; i < BDIM * HDIM; i += 256) {
    sCum[i] = 0.0f; sH[i] = 0.0f; sH16[i] = (_Float16)0.0f;
  }
  __syncthreads();

  for (int t = 0; t < TDIM; ++t) {
    for (int i = tid; i < BDIM * HDIM; i += 256) {
      float cv = sCum[i] + mbuf[(size_t)t * BDIM * HDIM + i];
      sCum[i] = cv;
      sX16[i] = (_Float16)cv;
    }
    __syncthreads();

#pragma unroll
    for (int uu = 0; uu < 6; ++uu) {
      int u = wave * 6 + uu;              // 0..47
      int isH = (u >= 24);
      int nt = isH ? (u - 24) : u;        // 0..23
      const _Float16* W = isH ? whh16 : wih16;
      const _Float16* X = isH ? sH16 : sX16;
      const float* bias = isH ? bhh : bih;
      _Float16* G = isH ? sGh : sGi;
      const int nabs = nt * 16 + m;
      v8f c = zero8();
#pragma unroll
      for (int kt = 0; kt < 4; ++kt) {
        v16h a = afrag_row(X + m * HDIM, kt, hi);
        v16h bfr = bfrag_nk(W + (size_t)nabs * HDIM, kt, hi);
        c = wmma16(a, bfr, c);
      }
      float bv = bias[nabs];
#pragma unroll
      for (int r = 0; r < 8; ++r) {
        int mr = r + (hi ? 8 : 0);
        G[mr * (3 * HDIM) + nabs] = (_Float16)(c[r] + bv);
      }
    }
    __syncthreads();

    for (int i = tid; i < BDIM * HDIM; i += 256) {
      int bb = i >> 7, h = i & 127;
      int gbase = bb * 3 * HDIM;
      float ir  = (float)sGi[gbase + h];
      float iz  = (float)sGi[gbase + HDIM + h];
      float inn = (float)sGi[gbase + 2 * HDIM + h];
      float hr  = (float)sGh[gbase + h];
      float hz  = (float)sGh[gbase + HDIM + h];
      float hn  = (float)sGh[gbase + 2 * HDIM + h];
      float rg = 1.0f / (1.0f + __expf(-(ir + hr)));
      float zg = 1.0f / (1.0f + __expf(-(iz + hz)));
      float ng = tanhf(inn + rg * hn);
      float hnew = (1.0f - zg) * ng + zg * sH[i];
      sH[i] = hnew;
      sH16[i] = (_Float16)hnew;
      rnn16[(size_t)t * BDIM * HDIM + i] = (_Float16)hnew;
    }
    __syncthreads();
  }
}

// -------------------- out[b,hn,t,:] = rnn[t,b,:] @ Wo + bo (broadcast over hn)
__global__ __launch_bounds__(32) void k_out(
    const _Float16* __restrict__ rnn16, const _Float16* __restrict__ woT,
    const float* __restrict__ bo, float* __restrict__ out)
{
  const int t = blockIdx.x;
  const int lane = threadIdx.x & 31;
  const int hi = (lane >> 4) & 1;
  const int m = lane & 15;

  v16h af[4];
#pragma unroll
  for (int kt = 0; kt < 4; ++kt)
    af[kt] = afrag_row(rnn16 + ((size_t)t * BDIM + m) * HDIM, kt, hi);

#pragma unroll
  for (int nt = 0; nt < 4; ++nt) {
    v8f c = zero8();
    const int nabs = nt * 16 + m;
#pragma unroll
    for (int kt = 0; kt < 4; ++kt)
      c = wmma16(af[kt], bfrag_nk(woT + (size_t)nabs * HDIM, kt, hi), c);
    float bv = bo[nabs];
#pragma unroll
    for (int r = 0; r < 8; ++r) {
      int b = r + (hi ? 8 : 0);
      float v = c[r] + bv;
      for (int hn = 0; hn < HN_; ++hn)
        out[(((size_t)b * HN_ + hn) * TDIM + t) * DOUT_ + nabs] = v;
    }
  }
}

// ----------------------------------------------------------------- launch
extern "C" void kernel_launch(void* const* d_in, const int* in_sizes, int n_in,
                              void* d_out, int out_size, void* d_ws, size_t ws_size,
                              hipStream_t stream)
{
  const float* x   = (const float*)d_in[0];
  const float* adj = (const float*)d_in[1];
  const float* Wp  = (const float*)d_in[2];
  const float* bp  = (const float*)d_in[3];
  const float* W1  = (const float*)d_in[4];
  const float* b1  = (const float*)d_in[5];
  const float* W2  = (const float*)d_in[6];
  const float* b2  = (const float*)d_in[7];
  const float* Wn1 = (const float*)d_in[8];
  const float* bn1 = (const float*)d_in[9];
  const float* Wn2 = (const float*)d_in[10];
  const float* bn2 = (const float*)d_in[11];
  const float* Wih = (const float*)d_in[12];
  const float* Whh = (const float*)d_in[13];
  const float* bih = (const float*)d_in[14];
  const float* bhh = (const float*)d_in[15];
  const float* Wo  = (const float*)d_in[16];
  const float* bo  = (const float*)d_in[17];
  float* out = (float*)d_out;

  char* ws = (char*)d_ws;
  size_t off = 0;
  auto alloc = [&](size_t bytes) -> char* {
    char* p = ws + off;
    off += (bytes + 255) & ~(size_t)255;
    return p;
  };

  const size_t NR = (size_t)TDIM * BDIM * VDIM;   // 65536 rows
  _Float16* h16   = (_Float16*)alloc(NR * HDIM * 2);
  _Float16* ai16  = (_Float16*)alloc(NR * HDIM * 2);   // reused as tmp after k_edge
  _Float16* bj16  = (_Float16*)alloc(NR * HDIM * 2);
  _Float16* agg16 = (_Float16*)alloc(NR * HDIM * 2);
  float*    mbuf  = (float*)alloc((size_t)TDIM * BDIM * HDIM * 4);
  _Float16* rnn16 = (_Float16*)alloc((size_t)TDIM * BDIM * HDIM * 2);
  _Float16* wpT   = (_Float16*)alloc(8192 * 2);
  _Float16* w1aT  = (_Float16*)alloc(16384 * 2);
  _Float16* w1bT  = (_Float16*)alloc(16384 * 2);
  _Float16* w2T   = (_Float16*)alloc(16384 * 2);
  _Float16* wn1T  = (_Float16*)alloc(16384 * 2);
  _Float16* wn2T  = (_Float16*)alloc(16384 * 2);
  _Float16* wih16 = (_Float16*)alloc(49152 * 2);
  _Float16* whh16 = (_Float16*)alloc(49152 * 2);
  _Float16* woT   = (_Float16*)alloc(8192 * 2);

  (void)in_sizes; (void)n_in; (void)out_size; (void)ws_size;

  k_pack<<<192, 256, 0, stream>>>(Wp, W1, W2, Wn1, Wn2, Wih, Whh, Wo,
                                  wpT, w1aT, w1bT, w2T, wn1T, wn2T,
                                  wih16, whh16, woT);
  k_proj<<<4096, 32, 0, stream>>>(x, bp, wpT, h16);
  k_aibj<<<4096, 64, 0, stream>>>(h16, w1aT, w1bT, b1, ai16, bj16);
  k_edge<<<4096, 128, 0, stream>>>(ai16, bj16, adj, w2T, b2, agg16);
  k_mlp1<<<4096, 32, 0, stream>>>(agg16, wn1T, bn1, ai16 /*tmp*/);
  k_mlp2_mean<<<512, 32, 0, stream>>>(ai16, wn2T, bn2, mbuf);
  k_gru<<<1, 256, 0, stream>>>(mbuf, wih16, whh16, bih, bhh, rnn16);
  k_out<<<32, 32, 0, stream>>>(rnn16, woT, bo, out);
}
